// ZoomTransformerBlock_26980984553617
// MI455X (gfx1250) — matrix operand order repaired and checked by reading, not compiled, not run on hardware
//
#include <hip/hip_runtime.h>
#include <hip/hip_bf16.h>
#include <math.h>

// ---------------------------------------------------------------------------
// Problem constants (from the reference)
// ---------------------------------------------------------------------------
#define BQ   2
#define NQ   4096
#define DQ   512
#define HQ   8
#define DHQ  64
#define FFQ  2048
#define MROWS (BQ*NQ)                 // 8192
#define SCALE_F 0.04419417382415922f  // 512^-0.5

typedef __bf16 bf16;
typedef __attribute__((ext_vector_type(16))) __bf16 v16bf;
typedef __attribute__((ext_vector_type(8)))  __bf16 v8bf;
typedef __attribute__((ext_vector_type(8)))  float  v8f;

static __device__ __forceinline__ v16bf mk16(v8bf lo, v8bf hi) {
    v16bf r;
#pragma unroll
    for (int i = 0; i < 8; ++i) { r[i] = lo[i]; r[i + 8] = hi[i]; }
    return r;
}

static __device__ __forceinline__ v8f wmma_bf16(v16bf a, v16bf b, v8f c) {
    return __builtin_amdgcn_wmma_f32_16x16x32_bf16(false, a, false, b,
                                                   (short)0, c, false, false);
}

static __device__ __forceinline__ float redmax16(float v) {
    v = fmaxf(v, __shfl_xor(v, 1, 32));
    v = fmaxf(v, __shfl_xor(v, 2, 32));
    v = fmaxf(v, __shfl_xor(v, 4, 32));
    v = fmaxf(v, __shfl_xor(v, 8, 32));
    return v;
}
static __device__ __forceinline__ float redsum16(float v) {
    v += __shfl_xor(v, 1, 32);
    v += __shfl_xor(v, 2, 32);
    v += __shfl_xor(v, 4, 32);
    v += __shfl_xor(v, 8, 32);
    return v;
}
static __device__ __forceinline__ float redsum32(float v) {
    v += __shfl_xor(v, 1, 32);
    v += __shfl_xor(v, 2, 32);
    v += __shfl_xor(v, 4, 32);
    v += __shfl_xor(v, 8, 32);
    v += __shfl_xor(v, 16, 32);
    return v;
}

// async global -> LDS copy of 16B per lane (CDNA5, tracked by ASYNCcnt)
static __device__ __forceinline__ void async_ld16(unsigned lds_off, const bf16* g) {
    asm volatile("global_load_async_to_lds_b128 %0, %1, off"
                 :: "v"(lds_off), "v"(g) : "memory");
}
static __device__ __forceinline__ unsigned lds_addr(const void* p) {
    return (unsigned)(unsigned long long)p;   // low 32 bits = LDS offset
}

// ---------------------------------------------------------------------------
// Weight prep: fp32 [R,C] row-major -> bf16 [C,R]
// ---------------------------------------------------------------------------
__global__ __launch_bounds__(256) void cast_transpose_kernel(
        const float* __restrict__ in, bf16* __restrict__ out, int R, int C) {
    int i = blockIdx.x * 256 + threadIdx.x;
    if (i >= R * C) return;
    int r = i / C, c = i % C;
    out[(size_t)c * R + r] = (bf16)in[i];
}

// ---------------------------------------------------------------------------
// LayerNorm (one wave per 512-elem row) -> bf16
// ---------------------------------------------------------------------------
__global__ __launch_bounds__(256) void ln_bf16_kernel(
        const float* __restrict__ x, const float* __restrict__ w,
        const float* __restrict__ b, bf16* __restrict__ out, int rows) {
    int row  = blockIdx.x * 8 + (threadIdx.x >> 5);
    int lane = threadIdx.x & 31;
    if (row >= rows) return;
    const float* xr = x + (size_t)row * DQ;
    float v[16]; float s = 0.f, s2 = 0.f;
#pragma unroll
    for (int i = 0; i < 16; ++i) {
        float t = xr[i * 32 + lane];
        v[i] = t; s += t; s2 += t * t;
    }
    s = redsum32(s); s2 = redsum32(s2);
    float mu = s * (1.f / DQ);
    float var = s2 * (1.f / DQ) - mu * mu;
    float rs = rsqrtf(var + 1e-5f);
    bf16* orow = out + (size_t)row * DQ;
#pragma unroll
    for (int i = 0; i < 16; ++i) {
        int idx = i * 32 + lane;
        orow[idx] = (bf16)((v[i] - mu) * rs * w[idx] + b[idx]);
    }
}

// ---------------------------------------------------------------------------
// bf16 WMMA GEMM:  C[M,Nw] = A[M,K] (row-major) * Bt[Nw,K]^T
// Block = 8 waves (4 along M, 2 along N); wave tile 32x64; block tile 128x128.
// Epilogues:
//   0: +bias, scatter q/k into [B,H,N,DH] and v TRANSPOSED into [B,H,DH,N]
//   1: +bias + fp32 residual(aux) -> fp32 out
//   2: +bias + exact GELU         -> bf16 out
//   3: +bias + fp32 residual(aux) -> fp32 out (final)
// ---------------------------------------------------------------------------
template <int KDIM, int EPI>
__global__ __launch_bounds__(256) void gemm_wmma_kernel(
        const bf16* __restrict__ A, const bf16* __restrict__ Bt,
        const float* __restrict__ bias, const float* __restrict__ aux,
        void* __restrict__ outp, int Nw) {
    const int wave = threadIdx.x >> 5;
    const int lane = threadIdx.x & 31;
    const int g    = lane >> 4;
    const int nl   = lane & 15;
    const int m0   = blockIdx.y * 128 + (wave & 3) * 32;
    const int n0   = blockIdx.x * 128 + (wave >> 2) * 64;

    const bf16* arow0 = A + (size_t)(m0 + nl) * KDIM;
    const bf16* arow1 = arow0 + (size_t)16 * KDIM;

    v8f acc[2][4];
#pragma unroll
    for (int rb = 0; rb < 2; ++rb)
#pragma unroll
        for (int t = 0; t < 4; ++t)
            acc[rb][t] = v8f{0.f,0.f,0.f,0.f,0.f,0.f,0.f,0.f};

    for (int k = 0; k < KDIM; k += 32) {
        v16bf af0 = *(const v16bf*)(arow0 + k + g * 16);
        v16bf af1 = *(const v16bf*)(arow1 + k + g * 16);
#pragma unroll
        for (int nt = 0; nt < 4; ++nt) {
            const bf16* bcol = Bt + (size_t)(n0 + nt * 16 + nl) * KDIM + k + g * 16;
            v16bf bfv = *(const v16bf*)bcol;
            acc[0][nt] = wmma_bf16(af0, bfv, acc[0][nt]);
            acc[1][nt] = wmma_bf16(af1, bfv, acc[1][nt]);
        }
    }

    // ---- epilogue: (m = m0 + rb*16 + r + 8*g,  n = n0 + nt*16 + nl) ----
    if constexpr (EPI == 0) {
        bf16* qkv = (bf16*)outp;
        const size_t BHND = (size_t)BQ * HQ * NQ * DHQ;
#pragma unroll
        for (int nt = 0; nt < 4; ++nt) {
            int j = n0 + nt * 16 + nl;
            float bs = bias[j];
            int which = j >> 9;          // 0=q 1=k 2=v
            int rem   = j & 511;
            int hh    = rem >> 6;
            int dh    = rem & 63;
#pragma unroll
            for (int rb = 0; rb < 2; ++rb)
#pragma unroll
            for (int r = 0; r < 8; ++r) {
                int m  = m0 + rb * 16 + r + 8 * g;
                int bb = m >> 12;
                int nn = m & 4095;
                int bhh = bb * HQ + hh;
                size_t idx;
                if (which == 2)  // v stored transposed: [B,H,DH,N]
                    idx = 2 * BHND + ((size_t)bhh * DHQ + dh) * NQ + nn;
                else
                    idx = (size_t)which * BHND + ((size_t)bhh * NQ + nn) * DHQ + dh;
                qkv[idx] = (bf16)(acc[rb][nt][r] + bs);
            }
        }
    } else if constexpr (EPI == 1 || EPI == 3) {
        float* o = (float*)outp;
#pragma unroll
        for (int nt = 0; nt < 4; ++nt) {
            int j = n0 + nt * 16 + nl;
            float bs = bias[j];
#pragma unroll
            for (int rb = 0; rb < 2; ++rb)
#pragma unroll
            for (int r = 0; r < 8; ++r) {
                int m = m0 + rb * 16 + r + 8 * g;
                size_t idx = (size_t)m * DQ + j;
                o[idx] = acc[rb][nt][r] + bs + aux[idx];
            }
        }
    } else { // EPI == 2: GELU -> bf16
        bf16* o = (bf16*)outp;
#pragma unroll
        for (int nt = 0; nt < 4; ++nt) {
            int j = n0 + nt * 16 + nl;
            float bs = bias[j];
#pragma unroll
            for (int rb = 0; rb < 2; ++rb)
#pragma unroll
            for (int r = 0; r < 8; ++r) {
                int m = m0 + rb * 16 + r + 8 * g;
                float v = acc[rb][nt][r] + bs;
                v = 0.5f * v * (1.f + erff(v * 0.70710678118654752f));
                o[(size_t)m * Nw + j] = (bf16)v;
            }
        }
    }
}

// ---------------------------------------------------------------------------
// Flash attention, Bc = 64 keys/step, double-buffered async K/V tiles.
// grid = (B*H)*(N/64) blocks, 4 waves * 16 query rows.
// K tiles are [key][dh] row-major; V is pre-transposed in global ([dh][n]) so
// both tiles are contiguous copies -> pure global_load_async_to_lds_b128.
// ---------------------------------------------------------------------------
__global__ __launch_bounds__(128) void attn_kernel(
        const bf16* __restrict__ qbuf, const bf16* __restrict__ kbuf,
        const bf16* __restrict__ vtbuf, const float* __restrict__ gamma,
        const float* __restrict__ dist, bf16* __restrict__ attn_out) {
    const int bh   = blockIdx.x >> 6;
    const int rb   = blockIdx.x & 63;
    const int b    = bh >> 3;
    const int h    = bh & 7;
    const int wave = threadIdx.x >> 5;
    const int lane = threadIdx.x & 31;
    const int g    = lane >> 4;
    const int nl   = lane & 15;
    const int qrow0 = rb * 64 + wave * 16;

    __shared__ __align__(32) bf16 kT[2][64 * 64];   // [key][dh]
    __shared__ __align__(32) bf16 vT[2][64 * 64];   // [dh][key]
    __shared__ __align__(32) bf16 pT[4][16 * 64];   // per-wave P tile

    const bf16* kbase  = kbuf  + (size_t)bh * NQ * DHQ;
    const bf16* vtbase = vtbuf + (size_t)bh * DHQ * NQ;

    // Q A-fragments (16 rows x 64 dh -> 2 k-steps), kept in registers
    const bf16* qb = qbuf + ((size_t)bh * NQ + qrow0 + nl) * DHQ;
    v16bf qa0 = *(const v16bf*)(qb + g * 16);
    v16bf qa1 = *(const v16bf*)(qb + 32 + g * 16);

    const float gma = gamma[b];

    // async tile issue: 8KB per matrix = 512 x 16B chunks; 128 thr x 4 chunks
    auto issue_tile = [&](int j0, int buf) {
        unsigned kofs = lds_addr(&kT[buf][0]);
        unsigned vofs = lds_addr(&vT[buf][0]);
#pragma unroll
        for (int i = 0; i < 4; ++i) {
            int c   = threadIdx.x + 128 * i;      // chunk id, 8 chunks per row
            int row = c >> 3;
            int seg = (c & 7) * 8;                // bf16 elems
            async_ld16(kofs + c * 16, kbase  + (size_t)(j0 + row) * DHQ + seg);
            async_ld16(vofs + c * 16, vtbase + (size_t)row * NQ + j0 + seg);
        }
    };

    v8f o[4];
#pragma unroll
    for (int t = 0; t < 4; ++t) o[t] = v8f{0.f,0.f,0.f,0.f,0.f,0.f,0.f,0.f};
    float mrun[8], lrun[8];
#pragma unroll
    for (int r = 0; r < 8; ++r) { mrun[r] = -1e30f; lrun[r] = 0.f; }

    constexpr int NIT = NQ / 64;
    issue_tile(0, 0);

    for (int it = 0; it < NIT; ++it) {
        const int cur = it & 1;
        if (it + 1 < NIT) {
            issue_tile((it + 1) * 64, (it + 1) & 1);
            asm volatile("s_wait_asynccnt 0x8" ::: "memory");  // cur tile done
        } else {
            asm volatile("s_wait_asynccnt 0x0" ::: "memory");
        }
        __syncthreads();

        const bf16* kt  = &kT[cur][0];
        const bf16* vtc = &vT[cur][0];
        const int   j0  = it * 64;

        // S = Q*K^T : 4 n-tiles x 2 k-steps
        v8f s[4];
#pragma unroll
        for (int nt = 0; nt < 4; ++nt) {
            s[nt] = v8f{0.f,0.f,0.f,0.f,0.f,0.f,0.f,0.f};
            v16bf kb;
            kb = *(const v16bf*)(kt + (size_t)(nt * 16 + nl) * 64 + g * 16);
            s[nt] = wmma_bf16(qa0, kb, s[nt]);
            kb = *(const v16bf*)(kt + (size_t)(nt * 16 + nl) * 64 + 32 + g * 16);
            s[nt] = wmma_bf16(qa1, kb, s[nt]);
        }

        // scale + distance bias
        float p[4][8];
#pragma unroll
        for (int r = 0; r < 8; ++r) {
            const float* dr = dist + (size_t)(qrow0 + r + 8 * g) * NQ + j0;
#pragma unroll
            for (int nt = 0; nt < 4; ++nt)
                p[nt][r] = s[nt][r] * SCALE_F - gma * dr[nt * 16 + nl];
        }

        // online softmax per row
#pragma unroll
        for (int r = 0; r < 8; ++r) {
            float mx = fmaxf(fmaxf(p[0][r], p[1][r]), fmaxf(p[2][r], p[3][r]));
            mx = redmax16(mx);
            float mnew  = fmaxf(mrun[r], mx);
            float alpha = __expf(mrun[r] - mnew);
            mrun[r] = mnew;
            float sl = 0.f;
#pragma unroll
            for (int nt = 0; nt < 4; ++nt) {
                p[nt][r] = __expf(p[nt][r] - mnew);
                sl += p[nt][r];
            }
            sl = redsum16(sl);
            lrun[r] = lrun[r] * alpha + sl;
#pragma unroll
            for (int nt = 0; nt < 4; ++nt) o[nt][r] *= alpha;
        }

        // C-layout fp32 P -> A-layout bf16 via per-wave LDS bounce
        bf16* pw = &pT[wave][0];
#pragma unroll
        for (int r = 0; r < 8; ++r) {
            int m = r + 8 * g;
#pragma unroll
            for (int nt = 0; nt < 4; ++nt)
                pw[m * 64 + nt * 16 + nl] = (bf16)p[nt][r];
        }
        asm volatile("s_wait_dscnt 0" ::: "memory");
        v16bf pa0 = *(const v16bf*)(pw + nl * 64 + g * 16);
        v16bf pa1 = *(const v16bf*)(pw + nl * 64 + 32 + g * 16);

        // O += P*V : 4 dh-tiles x 2 k-steps
#pragma unroll
        for (int nt = 0; nt < 4; ++nt) {
            v16bf vb;
            vb = *(const v16bf*)(vtc + (size_t)(nt * 16 + nl) * 64 + g * 16);
            o[nt] = wmma_bf16(pa0, vb, o[nt]);
            vb = *(const v16bf*)(vtc + (size_t)(nt * 16 + nl) * 64 + 32 + g * 16);
            o[nt] = wmma_bf16(pa1, vb, o[nt]);
        }
        __syncthreads();   // everyone done with buffers before next async write
    }

    // finalize: /rowsum, write [B,N,D] (D index = h*64+dh) bf16
#pragma unroll
    for (int r = 0; r < 8; ++r) {
        float inv = 1.0f / lrun[r];
        int qrow = qrow0 + r + 8 * g;
        bf16* dst = attn_out + ((size_t)b * NQ + qrow) * DQ + h * DHQ;
#pragma unroll
        for (int nt = 0; nt < 4; ++nt)
            dst[nt * 16 + nl] = (bf16)(o[nt][r] * inv);
    }
}

// ---------------------------------------------------------------------------
// Host-side launch
// ---------------------------------------------------------------------------
extern "C" void kernel_launch(void* const* d_in, const int* in_sizes, int n_in,
                              void* d_out, int out_size, void* d_ws, size_t ws_size,
                              hipStream_t stream) {
    const float* x      = (const float*)d_in[0];
    const float* gamma  = (const float*)d_in[1];
    const float* dist   = (const float*)d_in[2];
    const float* ln1_w  = (const float*)d_in[3];
    const float* ln1_b  = (const float*)d_in[4];
    const float* qkv_w  = (const float*)d_in[5];
    const float* qkv_b  = (const float*)d_in[6];
    const float* proj_w = (const float*)d_in[7];
    const float* proj_b = (const float*)d_in[8];
    const float* ln2_w  = (const float*)d_in[9];
    const float* ln2_b  = (const float*)d_in[10];
    const float* fc1_w  = (const float*)d_in[11];
    const float* fc1_b  = (const float*)d_in[12];
    const float* fc2_w  = (const float*)d_in[13];
    const float* fc2_b  = (const float*)d_in[14];
    float* out = (float*)d_out;

    char* wsb = (char*)d_ws;
    size_t off = 0;
    auto alloc = [&](size_t bytes) -> void* {
        void* p = wsb + off;
        off = (off + bytes + 255) & ~(size_t)255;
        return p;
    };
    const size_t BHND = (size_t)BQ * HQ * NQ * DHQ;
    bf16*  hb      = (bf16*) alloc((size_t)MROWS * DQ * 2);
    bf16*  qkvw_t  = (bf16*) alloc((size_t)3 * DQ * DQ * 2);
    bf16*  projw_t = (bf16*) alloc((size_t)DQ * DQ * 2);
    bf16*  fc1w_t  = (bf16*) alloc((size_t)FFQ * DQ * 2);
    bf16*  fc2w_t  = (bf16*) alloc((size_t)DQ * FFQ * 2);
    bf16*  qkvbuf  = (bf16*) alloc(3 * BHND * 2);     // q | k | v^T
    bf16*  attn_o  = (bf16*) alloc((size_t)MROWS * DQ * 2);
    float* x2      = (float*)alloc((size_t)MROWS * DQ * 4);
    bf16*  h2b     = (bf16*) alloc((size_t)MROWS * DQ * 2);
    bf16*  hmid    = (bf16*) alloc((size_t)MROWS * FFQ * 2);

    cast_transpose_kernel<<<(DQ * 3 * DQ + 255) / 256, 256, 0, stream>>>(qkv_w,  qkvw_t,  DQ,  3 * DQ);
    cast_transpose_kernel<<<(DQ * DQ + 255) / 256,     256, 0, stream>>>(proj_w, projw_t, DQ,  DQ);
    cast_transpose_kernel<<<(DQ * FFQ + 255) / 256,    256, 0, stream>>>(fc1_w,  fc1w_t,  DQ,  FFQ);
    cast_transpose_kernel<<<(FFQ * DQ + 255) / 256,    256, 0, stream>>>(fc2_w,  fc2w_t,  FFQ, DQ);

    ln_bf16_kernel<<<MROWS / 8, 256, 0, stream>>>(x, ln1_w, ln1_b, hb, MROWS);

    gemm_wmma_kernel<DQ, 0><<<dim3(3 * DQ / 128, MROWS / 128), 256, 0, stream>>>(
        hb, qkvw_t, qkv_b, nullptr, qkvbuf, 3 * DQ);

    attn_kernel<<<BQ * HQ * (NQ / 64), 128, 0, stream>>>(
        qkvbuf, qkvbuf + BHND, qkvbuf + 2 * BHND, gamma, dist, attn_o);

    gemm_wmma_kernel<DQ, 1><<<dim3(DQ / 128, MROWS / 128), 256, 0, stream>>>(
        attn_o, projw_t, proj_b, x, x2, DQ);

    ln_bf16_kernel<<<MROWS / 8, 256, 0, stream>>>(x2, ln2_w, ln2_b, h2b, MROWS);

    gemm_wmma_kernel<DQ, 2><<<dim3(FFQ / 128, MROWS / 128), 256, 0, stream>>>(
        h2b, fc1w_t, fc1_b, nullptr, hmid, FFQ);

    gemm_wmma_kernel<FFQ, 3><<<dim3(DQ / 128, MROWS / 128), 256, 0, stream>>>(
        hmid, fc2w_t, fc2_b, x2, out, DQ);

    (void)in_sizes; (void)n_in; (void)out_size; (void)ws_size;
}
